// ST_Graph_Denoise_Block_58239756534307
// MI455X (gfx1250) — matrix operand order repaired and checked by reading, not compiled, not run on hardware
//
#include <hip/hip_runtime.h>

// ---------------- problem constants ----------------
#define B_   8
#define C_   192        // channels
#define N_   1568       // T*H*W = 8*14*14
#define C2_  384        // 2*C
#define NT_  98         // N_/16
#define KP_  16         // padded neighbor count (duplicates; max() invariant)

typedef __attribute__((ext_vector_type(16))) _Float16 v16h;
typedef __attribute__((ext_vector_type(8)))  float    v8f;

// ---------------- hi/lo f16 split ----------------
static __device__ __forceinline__ void split2(float v, _Float16& h, _Float16& l) {
  h = (_Float16)v;
  l = (_Float16)(v - (float)h);
}

// D += A*B with A,B split hi/lo (drop lo*lo term): 3 chained WMMAs
static __device__ __forceinline__ v8f wmma3(v16h ah, v16h al, v16h bh, v16h bl, v8f acc) {
  acc = __builtin_amdgcn_wmma_f32_16x16x32_f16(false, ah, false, bh, (short)0, acc, false, false);
  acc = __builtin_amdgcn_wmma_f32_16x16x32_f16(false, ah, false, bl, (short)0, acc, false, false);
  acc = __builtin_amdgcn_wmma_f32_16x16x32_f16(false, al, false, bh, (short)0, acc, false, false);
  return acc;
}

// ---------------- fragment-swizzled storage offsets (in halfs) ----------------
// A fragment (16x32): lane = r + 16*hl, element e -> k = (e&7) + 8*hl + 16*(e>>3)
static __device__ __forceinline__ size_t aSwz(int frag, int r, int krem) {
  int hl = (krem >> 3) & 1;
  int e  = (krem & 7) | ((krem >> 4) << 3);
  return ((size_t)frag * 32 + (r | (hl << 4))) * 16 + e;
}
// B fragment (32x16): lane = c + 16*hl, element e -> k = (e&7) + 8*hl + 16*(e>>3)
static __device__ __forceinline__ size_t bSwz(int frag, int krem, int c) {
  int hl = (krem >> 3) & 1;
  int e  = (krem & 7) | ((krem >> 4) << 3);
  return ((size_t)frag * 32 + (c | (hl << 4))) * 16 + e;
}

// ---------------- kernel 1: xf = x + pos^T ----------------
__global__ __launch_bounds__(256) void k_prep_xf(const float* __restrict__ x,
                                                 const float* __restrict__ pos,
                                                 float* __restrict__ xf) {
  int i = blockIdx.x * 256 + threadIdx.x;
  if (i >= B_ * C_ * N_) return;
  int n = i % N_;
  int c = (i / N_) % C_;
  xf[i] = x[i] + pos[n * C_ + c];
}

// ---------------- kernel 2: swizzle weights into A-frag hi/lo planes ----------------
__global__ __launch_bounds__(256) void k_swz_w(const float* __restrict__ We,
                                               const float* __restrict__ W1,
                                               const float* __restrict__ W2,
                                               _Float16* __restrict__ WeH, _Float16* __restrict__ WeL,
                                               _Float16* __restrict__ W1H, _Float16* __restrict__ W1L,
                                               _Float16* __restrict__ W2H, _Float16* __restrict__ W2L) {
  const int SWe = C2_ * C2_, SW1 = C_ * C2_, SW2 = C_ * C_;
  int i = blockIdx.x * 256 + threadIdx.x;
  const float* src; _Float16 *H, *L; int K, idx;
  if (i < SWe)                  { src = We; H = WeH; L = WeL; K = C2_; idx = i; }
  else if (i < SWe + SW1)       { src = W1; H = W1H; L = W1L; K = C2_; idx = i - SWe; }
  else if (i < SWe + SW1 + SW2) { src = W2; H = W2H; L = W2L; K = C_;  idx = i - SWe - SW1; }
  else return;
  int row = idx / K, col = idx % K;
  int frag = (row >> 4) * (K >> 5) + (col >> 5);
  _Float16 h, l; split2(src[idx], h, l);
  size_t off = aSwz(frag, row & 15, col & 31);
  H[off] = h; L[off] = l;
}

// ---------------- kernel 3: normalize over C; write P in A- and B-swizzled planes ----------------
__global__ __launch_bounds__(C_) void k_norm(const float* __restrict__ xf,
                                             _Float16* __restrict__ PAH, _Float16* __restrict__ PAL,
                                             _Float16* __restrict__ PBH, _Float16* __restrict__ PBL) {
  int bn = blockIdx.x;            // b*N_ + n
  int b = bn / N_, n = bn % N_;
  int c = threadIdx.x;
  float v = xf[((size_t)b * C_ + c) * N_ + n];
  float s = v * v;
  s += __shfl_xor(s, 1); s += __shfl_xor(s, 2); s += __shfl_xor(s, 4);
  s += __shfl_xor(s, 8); s += __shfl_xor(s, 16);
  __shared__ float part[C_ / 32];
  if ((c & 31) == 0) part[c >> 5] = s;
  __syncthreads();
  float tot = 0.f;
  #pragma unroll
  for (int w = 0; w < C_ / 32; ++w) tot += part[w];
  float inv = 1.0f / fmaxf(sqrtf(tot), 1e-12f);
  _Float16 h, l; split2(v * inv, h, l);
  int frag = (n >> 4) * 6 + (c >> 5);
  size_t base = (size_t)b * N_ * C_;
  size_t oa = base + aSwz(frag, n & 15, c & 31);   // rows of gram
  size_t ob = base + bSwz(frag, c & 31, n & 15);   // columns of gram
  PAH[oa] = h; PAL[oa] = l;
  PBH[ob] = h; PBL[ob] = l;
}

// ---------------- kernel 4: gram via WMMA + stable top-9 ----------------
__global__ __launch_bounds__(32) void k_gram_topk(const v16h* __restrict__ PAH, const v16h* __restrict__ PAL,
                                                  const v16h* __restrict__ PBH, const v16h* __restrict__ PBL,
                                                  int* __restrict__ nnIdx) {
  int b = blockIdx.x / NT_;
  int ntile = blockIdx.x % NT_;
  int n0 = ntile * 16;
  int lane = threadIdx.x;
  int hl = lane >> 4;
  size_t pbase = (size_t)b * (NT_ * 6 * 32);   // v16h units per batch

  v16h aH[6], aL[6];
  #pragma unroll
  for (int kk = 0; kk < 6; ++kk) {
    size_t idx = pbase + (size_t)(ntile * 6 + kk) * 32 + lane;
    aH[kk] = PAH[idx]; aL[kk] = PAL[idx];
  }

  float topv[9]; int topi[9];
  #pragma unroll
  for (int q = 0; q < 9; ++q) { topv[q] = -1e30f; topi[q] = 0; }

  __shared__ float sim[16][17];

  for (int mt = 0; mt < NT_; ++mt) {
    int m0 = mt * 16;
    v8f acc = {};
    #pragma unroll
    for (int kk = 0; kk < 6; ++kk) {
      size_t idx = pbase + (size_t)(mt * 6 + kk) * 32 + lane;
      v16h bh = PBH[idx], bl = PBL[idx];
      acc = wmma3(aH[kk], aL[kk], bh, bl, acc);
    }
    #pragma unroll
    for (int j = 0; j < 8; ++j) sim[j + 8 * hl][lane & 15] = acc[j];
    __syncthreads();
    if (lane < 16) {                        // lane owns point row n0+lane
      #pragma unroll
      for (int cc = 0; cc < 16; ++cc) {
        float v = sim[lane][cc];
        int   m = m0 + cc;
        if (v > topv[8]) {                  // strict > : ties keep earlier index (top_k semantics)
          topv[8] = v; topi[8] = m;
          #pragma unroll
          for (int t = 8; t > 0; --t) {
            if (topv[t] > topv[t - 1]) {
              float tv = topv[t]; topv[t] = topv[t - 1]; topv[t - 1] = tv;
              int   ti = topi[t]; topi[t] = topi[t - 1]; topi[t - 1] = ti;
            }
          }
        }
      }
    }
    __syncthreads();
  }
  if (lane < 16) {
    int n = n0 + lane;
    #pragma unroll
    for (int q = 0; q < KP_; ++q)           // pad 9..15 with last neighbor (max-invariant)
      nnIdx[((size_t)b * N_ + n) * KP_ + q] = topi[q < 9 ? q : 8];
  }
}

// ---------------- kernel 5: EdgeConv GEMM + relu + max_k ----------------
__global__ __launch_bounds__(256) void k_edge(const float* __restrict__ xf,
                                              const v16h* __restrict__ WeH, const v16h* __restrict__ WeL,
                                              const float* __restrict__ be,
                                              const int* __restrict__ nnIdx,
                                              _Float16* __restrict__ gH, _Float16* __restrict__ gL) {
  int b  = blockIdx.x / (N_ / 2);
  int n0 = (blockIdx.x % (N_ / 2)) * 2;     // 2 points per block -> 32 columns
  int tid = threadIdx.x;

  __shared__ v16h fH[24 * 32];              // 2 col-tiles x 12 k-steps, B-swizzled (24KB)
  __shared__ v16h fL[24 * 32];
  __shared__ int idxL[32];

  if (tid < 32) idxL[tid] = nnIdx[((size_t)b * N_ + n0 + (tid >> 4)) * KP_ + (tid & 15)];
  __syncthreads();

  _Float16* fHs = (_Float16*)fH;
  _Float16* fLs = (_Float16*)fL;
  const float* xfb = xf + (size_t)b * C_ * N_;
  for (int i = tid; i < C2_ * 32; i += 256) {
    int cch = i >> 5, col = i & 31;         // cch = feature row (K), col = 2 points x 16 k
    int n = n0 + (col >> 4);
    float v;
    if (cch < C_) v = xfb[(size_t)cch * N_ + n];                   // x_i (broadcast over k)
    else {
      int cc = cch - C_;
      int m  = idxL[col];
      v = xfb[(size_t)cc * N_ + m] - xfb[(size_t)cc * N_ + n];     // x_j - x_i
    }
    _Float16 h, l; split2(v, h, l);
    size_t off = bSwz((col >> 4) * 12 + (cch >> 5), cch & 31, col & 15);
    fHs[off] = h; fLs[off] = l;
  }
  __syncthreads();

  int wave = tid >> 5;
  int lane = tid & 31;
  int cl = lane & 15, hl = lane >> 4;

  for (int t = wave; t < 24 * 2; t += 8) {  // 24 o-tiles x 2 col-tiles
    int ot = t >> 1, ct = t & 1;
    v8f acc = {};
    #pragma unroll
    for (int kk = 0; kk < 12; ++kk) {
      v16h ah = WeH[(ot * 12 + kk) * 32 + lane];
      v16h al = WeL[(ot * 12 + kk) * 32 + lane];
      v16h bh = fH[(ct * 12 + kk) * 32 + lane];
      v16h bl = fL[(ct * 12 + kk) * 32 + lane];
      acc = wmma3(ah, al, bh, bl, acc);
    }
    int n = n0 + ct;
    #pragma unroll
    for (int j = 0; j < 8; ++j) {
      int o = ot * 16 + j + 8 * hl;
      float v = fmaxf(acc[j] + be[o], 0.0f);        // relu(e)
      v = fmaxf(v, __shfl_xor(v, 1));               // max over 16 k-columns
      v = fmaxf(v, __shfl_xor(v, 2));               // (xor of bits 0..3 stays in half)
      v = fmaxf(v, __shfl_xor(v, 4));
      v = fmaxf(v, __shfl_xor(v, 8));
      if (cl == 0) {                                // write g into B-swizzled planes for W1 GEMM
        _Float16 h, l; split2(v, h, l);
        size_t off = (size_t)b * (C2_ * N_) + bSwz((n >> 4) * 12 + (o >> 5), o & 31, n & 15);
        gH[off] = h; gL[off] = l;
      }
    }
  }
}

// ---------------- kernel 6: h = relu(bn1(W1*g + b1)) ----------------
__global__ __launch_bounds__(128) void k_w1(const v16h* __restrict__ gHv, const v16h* __restrict__ gLv,
                                            const v16h* __restrict__ W1H, const v16h* __restrict__ W1L,
                                            const float* __restrict__ b1,
                                            const float* __restrict__ bn1g,
                                            const float* __restrict__ bn1b,
                                            float* __restrict__ hBuf,
                                            _Float16* __restrict__ hH, _Float16* __restrict__ hL) {
  int b  = blockIdx.x / NT_;
  int ntile = blockIdx.x % NT_;
  int n0 = ntile * 16;
  int wave = threadIdx.x >> 5;
  int lane = threadIdx.x & 31;
  int cl = lane & 15, hl = lane >> 4;
  size_t gbase = (size_t)b * (NT_ * 12 * 32);   // v16h units per batch
  const float bnscale = 1.0f / sqrtf(1.0f + 1e-5f);

  for (int ot = wave; ot < C_ / 16; ot += 4) {  // 12 o-tiles over 4 waves
    v8f acc = {};
    #pragma unroll
    for (int kk = 0; kk < 12; ++kk) {
      v16h ah = W1H[(ot * 12 + kk) * 32 + lane];
      v16h al = W1L[(ot * 12 + kk) * 32 + lane];
      size_t bi = gbase + (size_t)(ntile * 12 + kk) * 32 + lane;
      v16h bh = gHv[bi], bl = gLv[bi];
      acc = wmma3(ah, al, bh, bl, acc);
    }
    int n = n0 + cl;
    #pragma unroll
    for (int j = 0; j < 8; ++j) {
      int o = ot * 16 + j + 8 * hl;
      float v = (acc[j] + b1[o]) * (bn1g[o] * bnscale) + bn1b[o];
      v = fmaxf(v, 0.0f);
      hBuf[((size_t)b * C_ + o) * N_ + n] = v;              // f32 shortcut
      _Float16 h, l; split2(v, h, l);
      size_t off = (size_t)b * (C_ * N_) + bSwz((n >> 4) * 6 + (o >> 5), o & 31, n & 15);
      hH[off] = h; hL[off] = l;                             // B-swizzled for W2 GEMM
    }
  }
}

// ---------------- kernel 7: out = relu(bn2(W2*h + b2)) + h ----------------
__global__ __launch_bounds__(128) void k_w2(const v16h* __restrict__ hHv, const v16h* __restrict__ hLv,
                                            const v16h* __restrict__ W2H, const v16h* __restrict__ W2L,
                                            const float* __restrict__ b2,
                                            const float* __restrict__ bn2g,
                                            const float* __restrict__ bn2b,
                                            const float* __restrict__ hBuf,
                                            float* __restrict__ out) {
  int b  = blockIdx.x / NT_;
  int ntile = blockIdx.x % NT_;
  int n0 = ntile * 16;
  int wave = threadIdx.x >> 5;
  int lane = threadIdx.x & 31;
  int cl = lane & 15, hl = lane >> 4;
  size_t hbase = (size_t)b * (NT_ * 6 * 32);    // v16h units per batch
  const float bnscale = 1.0f / sqrtf(1.0f + 1e-5f);

  for (int ot = wave; ot < C_ / 16; ot += 4) {
    v8f acc = {};
    #pragma unroll
    for (int kk = 0; kk < 6; ++kk) {
      v16h ah = W2H[(ot * 6 + kk) * 32 + lane];
      v16h al = W2L[(ot * 6 + kk) * 32 + lane];
      size_t bi = hbase + (size_t)(ntile * 6 + kk) * 32 + lane;
      v16h bh = hHv[bi], bl = hLv[bi];
      acc = wmma3(ah, al, bh, bl, acc);
    }
    int n = n0 + cl;
    #pragma unroll
    for (int j = 0; j < 8; ++j) {
      int o = ot * 16 + j + 8 * hl;
      float v = (acc[j] + b2[o]) * (bn2g[o] * bnscale) + bn2b[o];
      v = fmaxf(v, 0.0f);
      size_t idx = ((size_t)b * C_ + o) * N_ + n;
      out[idx] = v + hBuf[idx];               // + shortcut
    }
  }
}

// ---------------- host launcher ----------------
extern "C" void kernel_launch(void* const* d_in, const int* in_sizes, int n_in,
                              void* d_out, int out_size, void* d_ws, size_t ws_size,
                              hipStream_t stream) {
  const float* x    = (const float*)d_in[0];
  const float* pos  = (const float*)d_in[1];
  const float* We   = (const float*)d_in[2];
  const float* be   = (const float*)d_in[3];
  const float* W1   = (const float*)d_in[4];
  const float* b1   = (const float*)d_in[5];
  const float* bn1g = (const float*)d_in[6];
  const float* bn1b = (const float*)d_in[7];
  const float* W2   = (const float*)d_in[8];
  const float* b2   = (const float*)d_in[9];
  const float* bn2g = (const float*)d_in[10];
  const float* bn2b = (const float*)d_in[11];
  float* out = (float*)d_out;

  char* w = (char*)d_ws;
  size_t off = 0;
  auto alloc = [&](size_t bytes) -> void* {
    void* p = w + off;
    off += (bytes + 255) & ~(size_t)255;
    return p;
  };
  float*    xf  = (float*)   alloc((size_t)B_ * C_ * N_ * 4);
  _Float16* WeH = (_Float16*)alloc((size_t)C2_ * C2_ * 2);
  _Float16* WeL = (_Float16*)alloc((size_t)C2_ * C2_ * 2);
  _Float16* W1H = (_Float16*)alloc((size_t)C_ * C2_ * 2);
  _Float16* W1L = (_Float16*)alloc((size_t)C_ * C2_ * 2);
  _Float16* W2H = (_Float16*)alloc((size_t)C_ * C_ * 2);
  _Float16* W2L = (_Float16*)alloc((size_t)C_ * C_ * 2);
  _Float16* PAH = (_Float16*)alloc((size_t)B_ * N_ * C_ * 2);
  _Float16* PAL = (_Float16*)alloc((size_t)B_ * N_ * C_ * 2);
  _Float16* PBH = (_Float16*)alloc((size_t)B_ * N_ * C_ * 2);
  _Float16* PBL = (_Float16*)alloc((size_t)B_ * N_ * C_ * 2);
  int*      nnIdx = (int*)   alloc((size_t)B_ * N_ * KP_ * 4);
  _Float16* gH  = (_Float16*)alloc((size_t)B_ * C2_ * N_ * 2);
  _Float16* gL  = (_Float16*)alloc((size_t)B_ * C2_ * N_ * 2);
  float*    hBuf= (float*)   alloc((size_t)B_ * C_ * N_ * 4);
  _Float16* hH  = (_Float16*)alloc((size_t)B_ * C_ * N_ * 2);
  _Float16* hL  = (_Float16*)alloc((size_t)B_ * C_ * N_ * 2);

  k_prep_xf<<<dim3((B_ * C_ * N_ + 255) / 256), dim3(256), 0, stream>>>(x, pos, xf);
  k_swz_w<<<dim3((C2_ * C2_ + C_ * C2_ + C_ * C_ + 255) / 256), dim3(256), 0, stream>>>(
      We, W1, W2, WeH, WeL, W1H, W1L, W2H, W2L);
  k_norm<<<dim3(B_ * N_), dim3(C_), 0, stream>>>(xf, PAH, PAL, PBH, PBL);
  k_gram_topk<<<dim3(B_ * NT_), dim3(32), 0, stream>>>(
      (const v16h*)PAH, (const v16h*)PAL, (const v16h*)PBH, (const v16h*)PBL, nnIdx);
  k_edge<<<dim3(B_ * (N_ / 2)), dim3(256), 0, stream>>>(
      xf, (const v16h*)WeH, (const v16h*)WeL, be, nnIdx, gH, gL);
  k_w1<<<dim3(B_ * NT_), dim3(128), 0, stream>>>(
      (const v16h*)gH, (const v16h*)gL, (const v16h*)W1H, (const v16h*)W1L,
      b1, bn1g, bn1b, hBuf, hH, hL);
  k_w2<<<dim3(B_ * NT_), dim3(128), 0, stream>>>(
      (const v16h*)hH, (const v16h*)hL, (const v16h*)W2H, (const v16h*)W2L,
      b2, bn2g, bn2b, hBuf, out);
}
